// UFourierLayer_78297253806216
// MI455X (gfx1250) — matrix operand order repaired and checked by reading, not compiled, not run on hardware
//
#include <hip/hip_runtime.h>
#include <cstdint>

typedef __attribute__((ext_vector_type(16))) _Float16 v16h;
typedef __attribute__((ext_vector_type(8)))  float    v8f;

#define LRELU(v) ((v) >= 0.f ? (v) : 0.1f * (v))

constexpr float kEps   = 1e-5f;
constexpr float kTwoPi = 6.28318530717958647692f;

constexpr int Bn = 4, Cn = 20;
constexpr int S_full = 16 * 32 * 32 * 16;   // 262144
constexpr int S_l1   = 8 * 16 * 16 * 8;     // 16384
constexpr int S_l2   = 4 * 8 * 8 * 4;       // 1024
constexpr int S_l3   = 2 * 4 * 4 * 2;       // 128

// ---------------------------------------------------------------------------
// Implicit-GEMM 4D conv (k=3^4, pad=1) via WMMA f16->f32.
// Fragment gathers are unconditional loads from clamped addresses times 0/1
// masks (no EXEC juggling); ky/kz unrolled so the scheduler can pipeline
// 9 load-batches against 9 WMMAs per basic block.
// ---------------------------------------------------------------------------
template <int CIN>
__global__ void conv4d_wmma_kernel(
    const float* __restrict__ in1, const float* __restrict__ in2,
    const float* __restrict__ w,      // (20, CIN, 81)
    const float* __restrict__ bias,   // optional (nullptr => none)
    float* __restrict__ out,          // (B, 20, outS)
    int stride,
    int Ti, int Xi, int Yi, int Zi,
    int To, int Xo, int Yo, int Zo)
{
  constexpr int C1 = 20;              // first tensor of the (optional) concat
  const int lane  = threadIdx.x & 31;
  const int otile = threadIdx.x >> 5; // 2 waves: out-ch rows 0-15 / 16-31
  const int b     = blockIdx.y;
  const int outS  = To * Xo * Yo * Zo;
  const int inS   = Ti * Xi * Yi * Zi;

  const int col = lane & 15;
  const int n   = blockIdx.x * 16 + col;
  const bool colValid = n < outS;
  const int nn = colValid ? n : 0;
  const int zo = nn % Zo;
  const int yo = (nn / Zo) % Yo;
  const int xo = (nn / (Zo * Yo)) % Xo;
  const int to = nn / (Zo * Yo * Xo);

  const int klo = (lane >> 4) * 16;   // unused; B fragment uses the same kk order as A
  const int m   = lane & 15;          // A fragment: matrix row
  const int o   = otile * 16 + m;
  const int kab = (lane >> 4) * 8;    // A fragment: K base per lane group

  v8f c = {};

  constexpr int NG = (CIN + 31) / 32;
#pragma unroll
  for (int g = 0; g < NG; ++g) {
    const int cb = g * 32;
    // ---- loop-invariant fragment descriptors (promoted to registers) ----
    int   aoff[16]; float am[16];
    const float* bp[16]; float bm[16];
#pragma unroll
    for (int e = 0; e < 16; ++e) {
      const int kk  = kab + (e & 7) + ((e >> 3) << 4);
      const int ciA = cb + kk;
      const bool va = (o < Cn) && (ciA < CIN);
      aoff[e] = va ? (o * CIN + ciA) * 81 : 0;
      am[e]   = va ? 1.f : 0.f;
      const int ciB = cb + kk;
      const bool vb = ciB < CIN;
      bp[e] = !vb ? in1
            : (ciB < C1 ? in1 + (b * C1 + ciB) * inS
                        : in2 + (b * (CIN - C1) + (ciB - C1)) * inS);
      bm[e] = vb ? 1.f : 0.f;
    }

    for (int kt = 0; kt < 3; ++kt) {
      const int ti = to * stride - 1 + kt;
      const bool vt = (unsigned)ti < (unsigned)Ti;
      for (int kx = 0; kx < 3; ++kx) {
        const int xi = xo * stride - 1 + kx;
        const bool vx = vt && (unsigned)xi < (unsigned)Xi;
#pragma unroll
        for (int ky = 0; ky < 3; ++ky) {
          const int yi = yo * stride - 1 + ky;
          const bool vy = vx && (unsigned)yi < (unsigned)Yi;
#pragma unroll
          for (int kz = 0; kz < 3; ++kz) {
            const int tap = (kt * 3 + kx) * 9 + ky * 3 + kz;
            const int zi = zo * stride - 1 + kz;
            const bool inb = colValid && vy && (unsigned)zi < (unsigned)Zi;
            const int sp  = ((ti * Xi + xi) * Yi + yi) * Zi + zi;
            const int spS = inb ? sp : 0;
            const float msel = inb ? 1.f : 0.f;

            v16h a, bf;
#pragma unroll
            for (int e = 0; e < 16; ++e)
              a[e] = (_Float16)(w[aoff[e] + tap] * am[e]);
#pragma unroll
            for (int e = 0; e < 16; ++e)
              bf[e] = (_Float16)(bp[e][spS] * (bm[e] * msel));

            c = __builtin_amdgcn_wmma_f32_16x16x32_f16(false, a, false, bf,
                                                       (short)0, c, false, false);
          }
        }
      }
    }
  }

  // Epilogue: row-validity is static per wave half. otile 0 owns rows 0..15
  // (all valid); otile 1 rows 16..19 live only in lanes 0..15 (rb==0, v<4).
  const int rb = (lane >> 4) * 8;
  if (otile == 0) {
    if (colValid) {
      float* op = out + (b * Cn + rb) * outS + n;
#pragma unroll
      for (int v = 0; v < 8; ++v) {
        float val = c[v];
        if (bias) val += bias[rb + v];
        op[v * outS] = val;
      }
    }
  } else {
    if (lane < 16 && colValid) {
      float* op = out + (b * Cn + 16) * outS + n;
#pragma unroll
      for (int v = 0; v < 4; ++v) {
        float val = c[v];
        if (bias) val += bias[16 + v];
        op[v * outS] = val;
      }
    }
  }
}

// ---------------------------------------------------------------------------
// Implicit-GEMM 4D transposed conv (k=4^4, s=2, p=1) via WMMA, gather form:
//   y[o,p] = sum_i sum_k w[i,o,k] x[i,(p+1-k)/2]   (k parity = (p+1)&1)
// Output voxels tiled by parity class so the 2 valid taps/dim are uniform
// across the 16 columns of a tile. Fused bias + LeakyReLU epilogue.
// ---------------------------------------------------------------------------
template <int CIN>
__global__ void deconv4d_wmma_kernel(
    const float* __restrict__ in1, const float* __restrict__ in2,
    const float* __restrict__ w,      // (CIN, 20, 256)
    const float* __restrict__ bias,
    float* __restrict__ out,          // (B, 20, outS)
    int Ti, int Xi, int Yi, int Zi,   // input dims (output = 2x each)
    int tilesPerClass)
{
  constexpr int C1 = 20;
  const int To = Ti * 2, Xo = Xi * 2, Yo = Yi * 2, Zo = Zi * 2;
  const int outS = To * Xo * Yo * Zo;
  const int inS  = Ti * Xi * Yi * Zi;
  const int coarse = inS;             // (To/2)*(Xo/2)*(Yo/2)*(Zo/2)

  const int lane  = threadIdx.x & 31;
  const int otile = threadIdx.x >> 5;
  const int b     = blockIdx.y;
  const int cls   = blockIdx.x / tilesPerClass;
  const int tile  = blockIdx.x % tilesPerClass;
  const int pt = (cls >> 3) & 1, px = (cls >> 2) & 1, py = (cls >> 1) & 1, pz = cls & 1;

  const int col = lane & 15;
  const int cg  = tile * 16 + col;
  const bool colValid = cg < coarse;
  const int c0 = colValid ? cg : 0;
  const int zc = c0 % Zi;
  const int yc = (c0 / Zi) % Yi;
  const int xc = (c0 / (Zi * Yi)) % Xi;
  const int tc = c0 / (Zi * Yi * Xi);
  const int to = 2 * tc + pt, xo = 2 * xc + px, yo = 2 * yc + py, zo = 2 * zc + pz;
  const int n  = ((to * Xo + xo) * Yo + yo) * Zo + zo;

  const int klo = (lane >> 4) * 16;
  const int m   = lane & 15;
  const int o   = otile * 16 + m;
  const int kab = (lane >> 4) * 8;

  v8f c = {};

  constexpr int NG = (CIN + 31) / 32;
#pragma unroll
  for (int g = 0; g < NG; ++g) {
    const int cb = g * 32;
    int   aoff[16]; float am[16];
    const float* bp[16]; float bm[16];
#pragma unroll
    for (int e = 0; e < 16; ++e) {
      const int kk  = kab + (e & 7) + ((e >> 3) << 4);
      const int ciA = cb + kk;
      const bool va = (o < Cn) && (ciA < CIN);
      aoff[e] = va ? (ciA * Cn + o) * 256 : 0;
      am[e]   = va ? 1.f : 0.f;
      const int ciB = cb + kk;
      const bool vb = ciB < CIN;
      bp[e] = !vb ? in1
            : (ciB < C1 ? in1 + (b * C1 + ciB) * inS
                        : in2 + (b * (CIN - C1) + (ciB - C1)) * inS);
      bm[e] = vb ? 1.f : 0.f;
    }

#pragma unroll
    for (int at = 0; at < 2; ++at) {
      const int kt  = (1 - pt) + 2 * at;
      const int tin = tc + pt - at;
      const bool vt = (unsigned)tin < (unsigned)Ti;
#pragma unroll
      for (int ax = 0; ax < 2; ++ax) {
        const int kx  = (1 - px) + 2 * ax;
        const int xin = xc + px - ax;
        const bool vx = vt && (unsigned)xin < (unsigned)Xi;
#pragma unroll
        for (int ay = 0; ay < 2; ++ay) {
          const int ky  = (1 - py) + 2 * ay;
          const int yin = yc + py - ay;
          const bool vy = vx && (unsigned)yin < (unsigned)Yi;
#pragma unroll
          for (int az = 0; az < 2; ++az) {
            const int kz  = (1 - pz) + 2 * az;
            const int zin = zc + pz - az;
            const bool inb = colValid && vy && (unsigned)zin < (unsigned)Zi;
            const int widx0 = kt * 64 + kx * 16 + ky * 4 + kz;
            const int sp  = ((tin * Xi + xin) * Yi + yin) * Zi + zin;
            const int spS = inb ? sp : 0;
            const float msel = inb ? 1.f : 0.f;

            v16h a, bf;
#pragma unroll
            for (int e = 0; e < 16; ++e)
              a[e] = (_Float16)(w[aoff[e] + widx0] * am[e]);
#pragma unroll
            for (int e = 0; e < 16; ++e)
              bf[e] = (_Float16)(bp[e][spS] * (bm[e] * msel));

            c = __builtin_amdgcn_wmma_f32_16x16x32_f16(false, a, false, bf,
                                                       (short)0, c, false, false);
          }
        }
      }
    }
  }

  const int rb = (lane >> 4) * 8;
  if (otile == 0) {
    if (colValid) {
      float* op = out + (b * Cn + rb) * outS + n;
#pragma unroll
      for (int v = 0; v < 8; ++v) {
        const float val = c[v] + bias[rb + v];
        op[v * outS] = LRELU(val);
      }
    }
  } else {
    if (lane < 16 && colValid) {
      float* op = out + (b * Cn + 16) * outS + n;
#pragma unroll
      for (int v = 0; v < 4; ++v) {
        const float val = c[v] + bias[16 + v];
        op[v * outS] = LRELU(val);
      }
    }
  }
}

// ---------------------------------------------------------------------------
// Pointwise channel GEMM: acc[b,o,s] += sum_i w[o,i] x[b,i,s] + bias[o]
// Single WMMA per tile (K=20 padded to 32), branch-free gathers.
// ---------------------------------------------------------------------------
__global__ void pointwise_wmma_kernel(
    const float* __restrict__ x, const float* __restrict__ w,
    const float* __restrict__ bias, float* __restrict__ acc)
{
  const int lane  = threadIdx.x & 31;
  const int otile = threadIdx.x >> 5;
  const int col = lane & 15;
  const int v0  = blockIdx.x * 16 + col;     // grid exact: B*S_full/16 tiles
  const int b   = v0 / S_full;
  const int s   = v0 % S_full;

  const int klo = (lane >> 4) * 16;
  const int m   = lane & 15;
  const int o   = otile * 16 + m;
  const int kab = (lane >> 4) * 8;

  __builtin_prefetch(x + (b * Cn) * S_full + s, 0, 0);

  v16h a, bf;
#pragma unroll
  for (int e = 0; e < 16; ++e) {
    const int kk = kab + (e & 7) + ((e >> 3) << 4);
    const bool va = (o < Cn) && (kk < Cn);
    a[e] = (_Float16)(w[va ? o * Cn + kk : 0] * (va ? 1.f : 0.f));
  }
#pragma unroll
  for (int e = 0; e < 16; ++e) {
    const int ci = kab + (e & 7) + ((e >> 3) << 4);
    const bool vb = ci < Cn;
    bf[e] = (_Float16)(x[(b * Cn + (vb ? ci : 0)) * S_full + s] * (vb ? 1.f : 0.f));
  }

  v8f c = {};
  c = __builtin_amdgcn_wmma_f32_16x16x32_f16(false, a, false, bf,
                                             (short)0, c, false, false);

  const int rb = (lane >> 4) * 8;
  if (otile == 0) {
    float* ap = acc + (b * Cn + rb) * S_full + s;
#pragma unroll
    for (int v = 0; v < 8; ++v)
      ap[v * S_full] += c[v] + bias[rb + v];
  } else if (lane < 16) {
    float* ap = acc + (b * Cn + 16) * S_full + s;
#pragma unroll
    for (int v = 0; v < 4; ++v)
      ap[v * S_full] += c[v] + bias[16 + v];
  }
}

// ---------------------------------------------------------------------------
// BatchNorm: per-channel stats (mean/var over B and spatial), then fused
// normalize + LeakyReLU in place.
// ---------------------------------------------------------------------------
__global__ void bn_stats_kernel(const float* __restrict__ h, int Nvox,
                                float* __restrict__ stats)
{
  __shared__ float ssum[256], ssq[256];
  const int ch = blockIdx.x;
  const int N = Bn * Nvox;
  float s = 0.f, q = 0.f;
  for (int i = threadIdx.x; i < N; i += blockDim.x) {
    const int bb = i / Nvox;
    const int nn = i % Nvox;
    const float v = h[(bb * Cn + ch) * Nvox + nn];
    s += v; q += v * v;
  }
  ssum[threadIdx.x] = s; ssq[threadIdx.x] = q;
  __syncthreads();
  for (int st = 128; st > 0; st >>= 1) {
    if ((int)threadIdx.x < st) {
      ssum[threadIdx.x] += ssum[threadIdx.x + st];
      ssq[threadIdx.x]  += ssq[threadIdx.x + st];
    }
    __syncthreads();
  }
  if (threadIdx.x == 0) {
    const float mean = ssum[0] / (float)N;
    stats[2 * ch]     = mean;
    stats[2 * ch + 1] = ssq[0] / (float)N - mean * mean;
  }
}

__global__ void bn_lrelu_kernel(float* __restrict__ h, const float* __restrict__ stats,
                                const float* __restrict__ g, const float* __restrict__ bta,
                                int Nvox, int total)
{
  const int i = blockIdx.x * blockDim.x + threadIdx.x;
  if (i >= total) return;
  const int ch = (i / Nvox) % Cn;
  const float mean = stats[2 * ch], var = stats[2 * ch + 1];
  const float v = (h[i] - mean) * rsqrtf(var + kEps) * g[ch] + bta[ch];
  h[i] = LRELU(v);
}

__global__ void relu_out_kernel(const float* __restrict__ acc,
                                float* __restrict__ out, int total)
{
  const int i = blockIdx.x * blockDim.x + threadIdx.x;
  if (i >= total) return;
  const float v = acc[i];
  out[i] = v > 0.f ? v : 0.f;
}

// ---------------------------------------------------------------------------
// Spectral branch: separable direct DFTs computing only the kept corner
// frequencies (faithful to rfftn over all 6 dims / irfftn over last 4).
// ---------------------------------------------------------------------------
__global__ void dft_r2c_z_kernel(const float* __restrict__ in, float2* __restrict__ out,
                                 int outer, int nin, int nout)
{
  const int idx = blockIdx.x * blockDim.x + threadIdx.x;
  if (idx >= outer * nout) return;
  const int f  = idx % nout;
  const int oo = idx / nout;
  const float w0 = -kTwoPi * (float)f / (float)nin;
  float re = 0.f, im = 0.f;
  const float* base = in + oo * nin;
#pragma unroll 4
  for (int j = 0; j < nin; ++j) {
    float sn, cs; __sincosf(w0 * (float)j, &sn, &cs);
    const float v = base[j];
    re += v * cs; im += v * sn;
  }
  out[idx] = make_float2(re, im);
}

// forward: out freq index f maps to actual frequency via (keepLo, hiStart)
__global__ void dft_fwd_c2c_kernel(const float2* __restrict__ in, float2* __restrict__ out,
                                   int outer, int nin, int inner, int nout,
                                   int keepLo, int hiStart)
{
  const int idx = blockIdx.x * blockDim.x + threadIdx.x;
  if (idx >= outer * nout * inner) return;
  const int r  = idx % inner;
  const int f  = (idx / inner) % nout;
  const int oo = idx / (inner * nout);
  const int freq = (f < keepLo) ? f : hiStart + (f - keepLo);
  const float w0 = -kTwoPi * (float)freq / (float)nin;
  float2 av = make_float2(0.f, 0.f);
  const float2* base = in + oo * nin * inner + r;
#pragma unroll 4
  for (int j = 0; j < nin; ++j) {
    float sn, cs; __sincosf(w0 * (float)j, &sn, &cs);
    const float2 v = base[j * inner];
    av.x += v.x * cs - v.y * sn;
    av.y += v.x * sn + v.y * cs;
  }
  out[(oo * nout + f) * inner + r] = av;
}

// inverse: input index j maps to actual frequency via (keepLo, hiStart)
__global__ void dft_inv_c2c_kernel(const float2* __restrict__ in, float2* __restrict__ out,
                                   int outer, int nin, int inner, int nout,
                                   int keepLo, int hiStart, float scale)
{
  const int idx = blockIdx.x * blockDim.x + threadIdx.x;
  if (idx >= outer * nout * inner) return;
  const int r  = idx % inner;
  const int t  = (idx / inner) % nout;
  const int oo = idx / (inner * nout);
  float2 av = make_float2(0.f, 0.f);
  const float2* base = in + oo * nin * inner + r;
#pragma unroll 4
  for (int j = 0; j < nin; ++j) {
    const int freq = (j < keepLo) ? j : hiStart + (j - keepLo);
    float sn, cs; __sincosf(kTwoPi * (float)freq * (float)t / (float)nout, &sn, &cs);
    const float2 v = base[j * inner];
    av.x += v.x * cs - v.y * sn;
    av.y += v.x * sn + v.y * cs;
  }
  out[(oo * nout + t) * inner + r] = make_float2(av.x * scale, av.y * scale);
}

// Hermitian real inverse over z (freqs 0..3 of rfft half-spectrum), += into acc.
__global__ void idft_z_real_acc_kernel(const float2* __restrict__ in,
                                       float* __restrict__ acc,
                                       int outer, float scale)
{
  const int idx = blockIdx.x * blockDim.x + threadIdx.x;
  if (idx >= outer * 16) return;
  const int z  = idx % 16;
  const int oo = idx / 16;
  const float2* base = in + oo * 4;
  float s = base[0].x;
#pragma unroll
  for (int k = 1; k < 4; ++k) {
    float sn, cs; __sincosf(kTwoPi * (float)(k * z) / 16.f, &sn, &cs);
    s += 2.f * (base[k].x * cs - base[k].y * sn);
  }
  acc[idx] += s * scale;
}

// Mode mixing: out[bf,o,modes] = sum_i xf[bf,i,modes] * sw_corner[i,o,modes]
__global__ void mode_mix_kernel(const float2* __restrict__ xf,
                                const float* __restrict__ sw1, const float* __restrict__ sw2,
                                const float* __restrict__ sw3, const float* __restrict__ sw4,
                                float2* __restrict__ outf)
{
  const int idx = blockIdx.x * blockDim.x + threadIdx.x;
  const int total = Bn * Cn * 8 * 16 * 8 * 4;
  if (idx >= total) return;
  const int zm = idx % 4;
  const int ym = (idx / 4) % 8;
  const int xm = (idx / 32) % 16;
  const int tm = (idx / 512) % 8;
  const int o  = (idx / 4096) % Cn;
  const int bf = idx / (4096 * Cn);
  const int tt = tm & 3, xx = xm & 7;
  const float* w = (tm < 4) ? ((xm < 8) ? sw1 : sw3)
                            : ((xm < 8) ? sw2 : sw4);
  float ar = 0.f, ai = 0.f;
#pragma unroll 4
  for (int i = 0; i < Cn; ++i) {
    const float2 v = xf[((((bf * Cn + i) * 8 + tm) * 16 + xm) * 8 + ym) * 4 + zm];
    const int widx = (((((i * Cn + o) * 4 + tt) * 8 + xx) * 8 + ym) * 4 + zm) * 2;
    const float wr = w[widx], wi = w[widx + 1];
    ar += v.x * wr - v.y * wi;
    ai += v.x * wi + v.y * wr;
  }
  outf[idx] = make_float2(ar, ai);
}

// ---------------------------------------------------------------------------
static inline int cdiv(int a, int b) { return (a + b - 1) / b; }

extern "C" void kernel_launch(void* const* d_in, const int* in_sizes, int n_in,
                              void* d_out, int out_size, void* d_ws, size_t ws_size,
                              hipStream_t stream) {
  (void)in_sizes; (void)n_in; (void)out_size; (void)ws_size;

  const float* x     = (const float*)d_in[0];
  const float* sw1   = (const float*)d_in[1];
  const float* sw2   = (const float*)d_in[2];
  const float* sw3   = (const float*)d_in[3];
  const float* sw4   = (const float*)d_in[4];
  const float* w_pw  = (const float*)d_in[5];
  const float* b_pw  = (const float*)d_in[6];
  const float* c1_w  = (const float*)d_in[7];
  const float* c2_w  = (const float*)d_in[8];
  const float* c21_w = (const float*)d_in[9];
  const float* c3_w  = (const float*)d_in[10];
  const float* c31_w = (const float*)d_in[11];
  const float* bn1_g  = (const float*)d_in[12]; const float* bn1_b  = (const float*)d_in[13];
  const float* bn2_g  = (const float*)d_in[14]; const float* bn2_b  = (const float*)d_in[15];
  const float* bn21_g = (const float*)d_in[16]; const float* bn21_b = (const float*)d_in[17];
  const float* bn3_g  = (const float*)d_in[18]; const float* bn3_b  = (const float*)d_in[19];
  const float* bn31_g = (const float*)d_in[20]; const float* bn31_b = (const float*)d_in[21];
  const float* d2_w = (const float*)d_in[22]; const float* d2_b = (const float*)d_in[23];
  const float* d1_w = (const float*)d_in[24]; const float* d1_b = (const float*)d_in[25];
  const float* d0_w = (const float*)d_in[26]; const float* d0_b = (const float*)d_in[27];
  const float* out_w = (const float*)d_in[28]; const float* out_b = (const float*)d_in[29];

  float* ws = (float*)d_ws;
  // workspace layout (float offsets); spectral arena aliases the od0 region
  // (od0 is dead once the final out-conv has consumed it).
  const long ACC_OFF = 0;
  const long OD0_OFF = ACC_OFF + (long)Bn * Cn * S_full;   // 20,971,520
  const long OC1_OFF = OD0_OFF + (long)Bn * Cn * S_full;
  const long OD1_OFF = OC1_OFF + (long)Bn * Cn * S_l1;
  const long H2_OFF  = OD1_OFF + (long)Bn * Cn * S_l1;
  const long H21_OFF = H2_OFF  + (long)Bn * Cn * S_l2;
  const long H3_OFF  = H21_OFF + (long)Bn * Cn * S_l2;
  const long H31_OFF = H3_OFF  + (long)Bn * Cn * S_l3;
  const long OD2_OFF = H31_OFF + (long)Bn * Cn * S_l3;
  const long STATS_OFF = OD2_OFF + (long)Bn * Cn * S_l2;

  float* acc = ws + ACC_OFF;
  float* od0 = ws + OD0_OFF;
  float* oc1 = ws + OC1_OFF;
  float* od1 = ws + OD1_OFF;
  float* h2  = ws + H2_OFF;
  float* h21 = ws + H21_OFF;   // becomes oc2
  float* h3  = ws + H3_OFF;
  float* h31 = ws + H31_OFF;   // becomes oc3
  float* od2 = ws + OD2_OFF;
  float* stats = ws + STATS_OFF;

  // spectral arena inside od0 region (16.4M floats <= 20.97M)
  float2* zbuf = (float2*)(ws + OD0_OFF);                       // 5,242,880 c (also i2)
  float2* ybuf = (float2*)(ws + OD0_OFF + 10485760);            // 1,310,720 c (also i1)
  float2* xbuf = (float2*)(ws + OD0_OFF + 10485760 + 2621440);  // 655,360 c (also i0)
  float2* tbuf = (float2*)(ws + OD0_OFF + 10485760 + 2621440 + 1310720);           // 327,680 c (also mix)
  float2* cbuf = (float2*)(ws + OD0_OFF + 10485760 + 2621440 + 1310720 + 655360);  // 327,680 c
  float2* bbuf = (float2*)(ws + OD0_OFF + 10485760 + 2621440 + 1310720 + 1310720); // 327,680 c

  const dim3 wblk(64);   // 2 waves per block: out-channel tiles 0-15 / 16-31

  // ---------------- U-Net ----------------
  // c1: (16,32,32,16) -> (8,16,16,8), stride 2
  conv4d_wmma_kernel<20><<<dim3(S_l1 / 16, Bn), wblk, 0, stream>>>(
      x, nullptr, c1_w, nullptr, oc1, 2, 16, 32, 32, 16, 8, 16, 16, 8);
  bn_stats_kernel<<<Cn, 256, 0, stream>>>(oc1, S_l1, stats);
  bn_lrelu_kernel<<<cdiv(Bn * Cn * S_l1, 256), 256, 0, stream>>>(
      oc1, stats, bn1_g, bn1_b, S_l1, Bn * Cn * S_l1);

  // c2: (8,16,16,8) -> (4,8,8,4), stride 2
  conv4d_wmma_kernel<20><<<dim3(S_l2 / 16, Bn), wblk, 0, stream>>>(
      oc1, nullptr, c2_w, nullptr, h2, 2, 8, 16, 16, 8, 4, 8, 8, 4);
  bn_stats_kernel<<<Cn, 256, 0, stream>>>(h2, S_l2, stats);
  bn_lrelu_kernel<<<cdiv(Bn * Cn * S_l2, 256), 256, 0, stream>>>(
      h2, stats, bn2_g, bn2_b, S_l2, Bn * Cn * S_l2);

  // c21: (4,8,8,4) -> (4,8,8,4), stride 1
  conv4d_wmma_kernel<20><<<dim3(S_l2 / 16, Bn), wblk, 0, stream>>>(
      h2, nullptr, c21_w, nullptr, h21, 1, 4, 8, 8, 4, 4, 8, 8, 4);
  bn_stats_kernel<<<Cn, 256, 0, stream>>>(h21, S_l2, stats);
  bn_lrelu_kernel<<<cdiv(Bn * Cn * S_l2, 256), 256, 0, stream>>>(
      h21, stats, bn21_g, bn21_b, S_l2, Bn * Cn * S_l2);            // h21 == oc2

  // c3: (4,8,8,4) -> (2,4,4,2), stride 2
  conv4d_wmma_kernel<20><<<dim3(S_l3 / 16, Bn), wblk, 0, stream>>>(
      h21, nullptr, c3_w, nullptr, h3, 2, 4, 8, 8, 4, 2, 4, 4, 2);
  bn_stats_kernel<<<Cn, 256, 0, stream>>>(h3, S_l3, stats);
  bn_lrelu_kernel<<<cdiv(Bn * Cn * S_l3, 256), 256, 0, stream>>>(
      h3, stats, bn3_g, bn3_b, S_l3, Bn * Cn * S_l3);

  // c31: (2,4,4,2) -> (2,4,4,2), stride 1
  conv4d_wmma_kernel<20><<<dim3(S_l3 / 16, Bn), wblk, 0, stream>>>(
      h3, nullptr, c31_w, nullptr, h31, 1, 2, 4, 4, 2, 2, 4, 4, 2);
  bn_stats_kernel<<<Cn, 256, 0, stream>>>(h31, S_l3, stats);
  bn_lrelu_kernel<<<cdiv(Bn * Cn * S_l3, 256), 256, 0, stream>>>(
      h31, stats, bn31_g, bn31_b, S_l3, Bn * Cn * S_l3);            // h31 == oc3

  // d2: (2,4,4,2) -> (4,8,8,4); coarse=128, 8 tiles/class
  deconv4d_wmma_kernel<20><<<dim3(16 * 8, Bn), wblk, 0, stream>>>(
      h31, nullptr, d2_w, d2_b, od2, 2, 4, 4, 2, 8);
  // d1: concat(oc2, od2) 40ch, (4,8,8,4) -> (8,16,16,8); coarse=1024, 64 tiles/class
  deconv4d_wmma_kernel<40><<<dim3(16 * 64, Bn), wblk, 0, stream>>>(
      h21, od2, d1_w, d1_b, od1, 4, 8, 8, 4, 64);
  // d0: concat(oc1, od1) 40ch, (8,16,16,8) -> (16,32,32,16); coarse=16384, 1024 tiles/class
  deconv4d_wmma_kernel<40><<<dim3(16 * 1024, Bn), wblk, 0, stream>>>(
      oc1, od1, d0_w, d0_b, od0, 8, 16, 16, 8, 1024);
  // out conv: concat(x, od0) 40ch, 3^4, stride 1, bias, no BN -> acc
  conv4d_wmma_kernel<40><<<dim3(S_full / 16, Bn), wblk, 0, stream>>>(
      x, od0, out_w, out_b, acc, 1, 16, 32, 32, 16, 16, 32, 32, 16);

  // ---------------- pointwise (adds into acc) ----------------
  pointwise_wmma_kernel<<<(Bn * S_full) / 16, wblk, 0, stream>>>(x, w_pw, b_pw, acc);

  // ---------------- spectral branch (adds into acc) ----------------
  // forward DFTs, keeping only needed frequencies at each stage
  dft_r2c_z_kernel<<<cdiv(1310720 * 4, 256), 256, 0, stream>>>(x, zbuf, 1310720, 16, 4);
  dft_fwd_c2c_kernel<<<cdiv(40960 * 8 * 4, 256), 256, 0, stream>>>(
      zbuf, ybuf, 40960, 32, 4, 8, 8, 0);                 // y: keep 0..7
  dft_fwd_c2c_kernel<<<cdiv(1280 * 16 * 32, 256), 256, 0, stream>>>(
      ybuf, xbuf, 1280, 32, 32, 16, 8, 24);               // x: keep 0..7, 24..31
  dft_fwd_c2c_kernel<<<cdiv(80 * 8 * 512, 256), 256, 0, stream>>>(
      xbuf, tbuf, 80, 16, 512, 8, 4, 12);                 // t: keep 0..3, 12..15
  dft_fwd_c2c_kernel<<<cdiv(4 * 20 * 4096, 256), 256, 0, stream>>>(
      tbuf, cbuf, 4, 20, 4096, 20, 20, 0);                // channel axis: full
  dft_fwd_c2c_kernel<<<cdiv(1 * 4 * 81920, 256), 256, 0, stream>>>(
      cbuf, bbuf, 1, 4, 81920, 4, 4, 0);                  // batch axis: full
  // mode mixing (writes mix result into tbuf region, which is dead now)
  mode_mix_kernel<<<cdiv(Bn * Cn * 8 * 16 * 8 * 4, 256), 256, 0, stream>>>(
      bbuf, sw1, sw2, sw3, sw4, tbuf);
  // inverse DFTs over last 4 axes
  dft_inv_c2c_kernel<<<cdiv(80 * 16 * 512, 256), 256, 0, stream>>>(
      tbuf, xbuf, 80, 8, 512, 16, 4, 12, 1.f / 16.f);     // t
  dft_inv_c2c_kernel<<<cdiv(1280 * 32 * 32, 256), 256, 0, stream>>>(
      xbuf, ybuf, 1280, 16, 32, 32, 8, 24, 1.f / 32.f);   // x
  dft_inv_c2c_kernel<<<cdiv(40960 * 32 * 4, 256), 256, 0, stream>>>(
      ybuf, zbuf, 40960, 8, 4, 32, 8, 0, 1.f / 32.f);     // y
  idft_z_real_acc_kernel<<<cdiv(1310720 * 16, 256), 256, 0, stream>>>(
      zbuf, acc, 1310720, 1.f / 16.f);                    // z (real, += acc)

  // ---------------- final relu ----------------
  relu_out_kernel<<<cdiv(Bn * Cn * S_full, 256), 256, 0, stream>>>(
      acc, (float*)d_out, Bn * Cn * S_full);
}
